// FP8Linear_37357625541065
// MI455X (gfx1250) — 8-bit matrix operand order repaired and checked by reading, compiled, not run on hardware
//
#include <hip/hip_runtime.h>
#include <hip/hip_bf16.h>

typedef int      v16i  __attribute__((ext_vector_type(16)));
typedef float    v8f   __attribute__((ext_vector_type(8)));
typedef unsigned uv4   __attribute__((ext_vector_type(4)));
typedef int      iv8   __attribute__((ext_vector_type(8)));
typedef int      iv4   __attribute__((ext_vector_type(4)));

#define K_DIM   4096
#define FP8MAX  448.0f

#if __has_builtin(__builtin_amdgcn_tensor_load_to_lds) && \
    __has_builtin(__builtin_amdgcn_s_wait_tensorcnt)
#define USE_TDM 1
#else
#define USE_TDM 0
#endif

__global__ void fp8lin_zero_amax(unsigned* amax_bits) {
    if (threadIdx.x == 0 && blockIdx.x == 0) amax_bits[0] = 0u;
}

__global__ void fp8lin_amax(const float* __restrict__ x, size_t n,
                            unsigned* __restrict__ amax_bits) {
    size_t tid    = (size_t)blockIdx.x * blockDim.x + threadIdx.x;
    size_t stride = (size_t)gridDim.x * blockDim.x;
    float m = 0.0f;
    for (size_t i = tid * 4; i < n; i += stride * 4) {
        float4 v = *(const float4*)(x + i);
        m = fmaxf(m, fmaxf(fmaxf(fabsf(v.x), fabsf(v.y)),
                           fmaxf(fabsf(v.z), fabsf(v.w))));
    }
    for (int off = 16; off > 0; off >>= 1)
        m = fmaxf(m, __shfl_xor(m, off, 32));
    __shared__ float smax[8];
    int lane = threadIdx.x & 31;
    int wave = threadIdx.x >> 5;
    if (lane == 0) smax[wave] = m;
    __syncthreads();
    if (wave == 0) {
        m = (lane < 8) ? smax[lane] : 0.0f;
        for (int off = 4; off > 0; off >>= 1)
            m = fmaxf(m, __shfl_xor(m, off, 32));
        if (lane == 0) atomicMax(amax_bits, __float_as_uint(m));
    }
}

__device__ inline unsigned char fp8lin_sw_cvt_e4m3(float x) {
    unsigned  u   = __float_as_uint(x);
    unsigned  sgn = (u >> 24) & 0x80u;
    float a = fminf(fabsf(x), FP8MAX);
    if (a < 0.001953125f * 0.0625f) return (unsigned char)sgn;
    unsigned bits = __float_as_uint(a);
    unsigned exp  = bits >> 23;
    unsigned mant = bits & 0x7FFFFFu;
    unsigned m3     = mant >> 20;
    unsigned lsb    = m3 & 1u;
    unsigned rb     = (mant >> 19) & 1u;
    unsigned sticky = (mant & 0x7FFFFu) ? 1u : 0u;
    m3 += (rb & (sticky | lsb));
    if (m3 == 8u) { m3 = 0u; exp += 1u; }
    int e8 = (int)exp - 127 + 7;
    if (e8 <= 0)  return (unsigned char)sgn;
    if (e8 > 15)  { e8 = 15; m3 = 6u; }
    return (unsigned char)(sgn | ((unsigned)e8 << 3) | m3);
}

__global__ void fp8lin_quant(const float* __restrict__ x,
                             unsigned char* __restrict__ xq,
                             const unsigned* __restrict__ amax_bits,
                             size_t n) {
    float amax = fmaxf(__uint_as_float(*amax_bits), 1e-12f);
    float sinv = FP8MAX / amax;
    size_t i = ((size_t)blockIdx.x * blockDim.x + threadIdx.x) * 16;
    if (i >= n) return;
    float4 v0 = *(const float4*)(x + i);
    float4 v1 = *(const float4*)(x + i + 4);
    float4 v2 = *(const float4*)(x + i + 8);
    float4 v3 = *(const float4*)(x + i + 12);
    uint4 o;
#if __has_builtin(__builtin_amdgcn_cvt_pk_fp8_f32)
    auto cl = [&](float f) { return fminf(fmaxf(f * sinv, -FP8MAX), FP8MAX); };
    int p;
    p = __builtin_amdgcn_cvt_pk_fp8_f32(cl(v0.x), cl(v0.y), 0, false);
    p = __builtin_amdgcn_cvt_pk_fp8_f32(cl(v0.z), cl(v0.w), p, true);
    o.x = (unsigned)p;
    p = __builtin_amdgcn_cvt_pk_fp8_f32(cl(v1.x), cl(v1.y), 0, false);
    p = __builtin_amdgcn_cvt_pk_fp8_f32(cl(v1.z), cl(v1.w), p, true);
    o.y = (unsigned)p;
    p = __builtin_amdgcn_cvt_pk_fp8_f32(cl(v2.x), cl(v2.y), 0, false);
    p = __builtin_amdgcn_cvt_pk_fp8_f32(cl(v2.z), cl(v2.w), p, true);
    o.z = (unsigned)p;
    p = __builtin_amdgcn_cvt_pk_fp8_f32(cl(v3.x), cl(v3.y), 0, false);
    p = __builtin_amdgcn_cvt_pk_fp8_f32(cl(v3.z), cl(v3.w), p, true);
    o.w = (unsigned)p;
#else
    float tmp[16] = {v0.x,v0.y,v0.z,v0.w, v1.x,v1.y,v1.z,v1.w,
                     v2.x,v2.y,v2.z,v2.w, v3.x,v3.y,v3.z,v3.w};
    unsigned b[16];
    for (int j = 0; j < 16; ++j) b[j] = fp8lin_sw_cvt_e4m3(tmp[j] * sinv);
    o.x = b[0] | (b[1] << 8) | (b[2] << 16) | (b[3] << 24);
    o.y = b[4] | (b[5] << 8) | (b[6] << 16) | (b[7] << 24);
    o.z = b[8] | (b[9] << 8) | (b[10] << 16) | (b[11] << 24);
    o.w = b[12] | (b[13] << 8) | (b[14] << 16) | (b[15] << 24);
#endif
    *(uint4*)(xq + i) = o;
}

#if USE_TDM
__device__ inline void tdm_load_2d(unsigned lds_byte_addr,
                                   const void* gaddr,
                                   unsigned tensor_rows,
                                   unsigned row_elems,
                                   unsigned tile_rows,
                                   unsigned tile_row_elems)
{
    unsigned long long ga = (unsigned long long)gaddr;
    uv4 g0;
    g0.x = 1u;
    g0.y = lds_byte_addr;
    g0.z = (unsigned)(ga & 0xFFFFFFFFull);
    g0.w = (unsigned)((ga >> 32) & 0x01FFFFFFull)
         | 0x80000000u;
    iv8 g1;
    g1[0] = (int)(2u << 16);
    g1[1] = (int)((row_elems & 0xFFFFu) << 16);
    g1[2] = (int)(((row_elems >> 16) & 0xFFFFu)
                | ((tensor_rows & 0xFFFFu) << 16));
    g1[3] = (int)(((tensor_rows >> 16) & 0xFFFFu)
                | ((tile_row_elems & 0xFFFFu) << 16));
    g1[4] = (int)(tile_rows & 0xFFFFu);
    g1[5] = (int)row_elems;
    g1[6] = 0;
    g1[7] = 0;
    iv4 z4 = {0, 0, 0, 0};
    iv8 z8 = {0, 0, 0, 0, 0, 0, 0, 0};
    __builtin_amdgcn_tensor_load_to_lds(g0, g1, z4, z4, z8, 0);
}
#endif

__global__ __launch_bounds__(256) void fp8lin_gemm(
    const unsigned char* __restrict__ A,
    const unsigned char* __restrict__ W,
    const float* __restrict__ wscale,
    const unsigned* __restrict__ amax_bits,
    const __hip_bfloat16* __restrict__ bias,
    __hip_bfloat16* __restrict__ out,
    int M, int N) {
    constexpr int K = K_DIM;
    constexpr unsigned A_BYTES = 64 * 128;
    constexpr unsigned B_BYTES = 128 * 128;
    constexpr unsigned BUF     = A_BYTES + B_BYTES;
    __shared__ __align__(16) unsigned char lds[2 * BUF];

    const int tid  = threadIdx.x;
    const int lane = tid & 31;
    const int wave = tid >> 5;
    const int wrow = wave >> 2;
    const int wcol = wave & 3;
    const int g    = lane >> 4;
    const int mrow = lane & 15;
    const int ncol = lane & 15;

    const int m0 = blockIdx.y * 64;
    const int n0 = blockIdx.x * 128;

    const float combined =
        (fmaxf(__uint_as_float(*amax_bits), 1e-12f) / FP8MAX) * wscale[0];

    v8f acc[2][2];
    #pragma unroll
    for (int i = 0; i < 2; ++i)
        #pragma unroll
        for (int j = 0; j < 2; ++j)
            #pragma unroll
            for (int e = 0; e < 8; ++e) acc[i][j][e] = 0.0f;

    const int nk = K / 128;

#if USE_TDM
    const unsigned lds_base = (unsigned)(unsigned long long)(const void*)&lds[0];

    if (wave == 0) {
        tdm_load_2d(lds_base,           A + (size_t)m0 * K, (unsigned)(M - m0),
                    K / 4, 64u, 32u);
        tdm_load_2d(lds_base + A_BYTES, W + (size_t)n0 * K, (unsigned)(N - n0),
                    K / 4, 128u, 32u);
        __builtin_amdgcn_s_wait_tensorcnt(0);
    }
    __syncthreads();
    asm volatile("" ::: "memory");
#else
    {
        const int ia0 = tid, ia1 = tid + 256;
        const int ib0 = tid, ib1 = tid + 256, ib2 = tid + 512, ib3 = tid + 768;
        uint4 ra0 = *(const uint4*)(A + (size_t)(m0 + (ia0 >> 3)) * K + (ia0 & 7) * 16);
        uint4 ra1 = *(const uint4*)(A + (size_t)(m0 + (ia1 >> 3)) * K + (ia1 & 7) * 16);
        uint4 rb0 = *(const uint4*)(W + (size_t)(n0 + (ib0 >> 3)) * K + (ib0 & 7) * 16);
        uint4 rb1 = *(const uint4*)(W + (size_t)(n0 + (ib1 >> 3)) * K + (ib1 & 7) * 16);
        uint4 rb2 = *(const uint4*)(W + (size_t)(n0 + (ib2 >> 3)) * K + (ib2 & 7) * 16);
        uint4 rb3 = *(const uint4*)(W + (size_t)(n0 + (ib3 >> 3)) * K + (ib3 & 7) * 16);
        *(uint4*)(lds + ia0 * 16) = ra0;
        *(uint4*)(lds + ia1 * 16) = ra1;
        *(uint4*)(lds + A_BYTES + ib0 * 16) = rb0;
        *(uint4*)(lds + A_BYTES + ib1 * 16) = rb1;
        *(uint4*)(lds + A_BYTES + ib2 * 16) = rb2;
        *(uint4*)(lds + A_BYTES + ib3 * 16) = rb3;
    }
    __syncthreads();
    uint4 ra0, ra1, rb0, rb1, rb2, rb3;
#endif

    int p = 0;
    for (int it = 0; it < nk; ++it) {
        const int k0 = it * 128;
        const bool more = (it + 1) < nk;

#if USE_TDM
        if (more && wave == 0) {
            const unsigned dst = lds_base + (unsigned)((p ^ 1) * BUF);
            tdm_load_2d(dst,           A + (size_t)m0 * K + k0 + 128,
                        (unsigned)(M - m0), K / 4, 64u, 32u);
            tdm_load_2d(dst + A_BYTES, W + (size_t)n0 * K + k0 + 128,
                        (unsigned)(N - n0), K / 4, 128u, 32u);
        }
#else
        if (more) {
            const int kn = k0 + 128;
            ra0 = *(const uint4*)(A + (size_t)(m0 + (tid >> 3)) * K + kn + (tid & 7) * 16);
            ra1 = *(const uint4*)(A + (size_t)(m0 + ((tid + 256) >> 3)) * K + kn + (tid & 7) * 16);
            rb0 = *(const uint4*)(W + (size_t)(n0 + (tid >> 3)) * K + kn + (tid & 7) * 16);
            rb1 = *(const uint4*)(W + (size_t)(n0 + ((tid + 256) >> 3)) * K + kn + (tid & 7) * 16);
            rb2 = *(const uint4*)(W + (size_t)(n0 + ((tid + 512) >> 3)) * K + kn + (tid & 7) * 16);
            rb3 = *(const uint4*)(W + (size_t)(n0 + ((tid + 768) >> 3)) * K + kn + (tid & 7) * 16);
        }
#endif

        const unsigned char* Ab = lds + p * BUF;
        const unsigned char* Bb = lds + p * BUF + A_BYTES;

        v16i a[2];
        #pragma unroll
        for (int i = 0; i < 2; ++i) {
            const unsigned char* ap =
                Ab + (wrow * 32 + i * 16 + mrow) * 128 + g * 16;
            #pragma unroll
            for (int d = 0; d < 16; d += 2) {
                int koff = ((d >> 2) * 32) + ((d & 2) ? 8 : 0);
                uint2 q = *(const uint2*)(ap + koff);
                a[i][d]     = (int)q.x;
                a[i][d + 1] = (int)q.y;
            }
        }
        v16i b[2];
        #pragma unroll
        for (int j = 0; j < 2; ++j) {
            const unsigned char* bp =
                Bb + (wcol * 32 + j * 16 + ncol) * 128 + g * 16;
            #pragma unroll
            for (int q4 = 0; q4 < 4; ++q4) {
                uint4 q = *(const uint4*)(bp + q4 * 32);
                b[j][4 * q4]     = (int)q.x;
                b[j][4 * q4 + 1] = (int)q.y;
                b[j][4 * q4 + 2] = (int)q.z;
                b[j][4 * q4 + 3] = (int)q.w;
            }
        }

        #pragma unroll
        for (int i = 0; i < 2; ++i)
            #pragma unroll
            for (int j = 0; j < 2; ++j)
                acc[i][j] = __builtin_amdgcn_wmma_f32_16x16x128_fp8_fp8(
                    a[i], b[j], (short)0, acc[i][j], false, false);

#if USE_TDM
        if (more && wave == 0) __builtin_amdgcn_s_wait_tensorcnt(0);
        __syncthreads();
        asm volatile("" ::: "memory");
#else
        __syncthreads();
        if (more) {
            unsigned char* D = lds + (p ^ 1) * BUF;
            *(uint4*)(D + tid * 16) = ra0;
            *(uint4*)(D + (tid + 256) * 16) = ra1;
            *(uint4*)(D + A_BYTES + tid * 16) = rb0;
            *(uint4*)(D + A_BYTES + (tid + 256) * 16) = rb1;
            *(uint4*)(D + A_BYTES + (tid + 512) * 16) = rb2;
            *(uint4*)(D + A_BYTES + (tid + 768) * 16) = rb3;
        }
        __syncthreads();
#endif
        p ^= 1;
    }

    #pragma unroll
    for (int i = 0; i < 2; ++i) {
        #pragma unroll
        for (int j = 0; j < 2; ++j) {
            int n = n0 + wcol * 32 + j * 16 + ncol;
            float bf = __bfloat162float(bias[n]);
            #pragma unroll
            for (int r = 0; r < 8; ++r) {
                int m = m0 + wrow * 32 + i * 16 + g * 8 + r;
                float v = acc[i][j][r] * combined;
                float o = __bfloat162float(__float2bfloat16(v)) + bf;
                out[(size_t)m * N + n] = __float2bfloat16(o);
            }
        }
    }
}

extern "C" void kernel_launch(void* const* d_in, const int* in_sizes, int n_in,
                              void* d_out, int out_size, void* d_ws, size_t ws_size,
                              hipStream_t stream) {
    const float*          x      = (const float*)d_in[0];
    const unsigned char*  w      = (const unsigned char*)d_in[1];
    const float*          wscale = (const float*)d_in[2];
    const __hip_bfloat16* bias   = (const __hip_bfloat16*)d_in[3];
    __hip_bfloat16*       out    = (__hip_bfloat16*)d_out;

    const int    K  = K_DIM;
    const size_t xn = (size_t)in_sizes[0];
    const int    M  = (int)(xn / (size_t)K);
    const int    N  = in_sizes[3];

    unsigned*      amax_bits = (unsigned*)d_ws;
    unsigned char* xq        = (unsigned char*)d_ws + 256;

    fp8lin_zero_amax<<<1, 32, 0, stream>>>(amax_bits);

    fp8lin_amax<<<2048, 256, 0, stream>>>(x, xn, amax_bits);

    size_t qthreads = xn / 16;
    fp8lin_quant<<<(unsigned)((qthreads + 255) / 256), 256, 0, stream>>>(
        x, xq, amax_bits, xn);

    dim3 grid(N / 128, M / 64);
    fp8lin_gemm<<<grid, 256, 0, stream>>>(xq, w, wscale, amax_bits, bias, out, M, N);
}
